// LDS_LR_35605278884573
// MI455X (gfx1250) — matrix operand order repaired and checked by reading, not compiled, not run on hardware
//
#include <hip/hip_runtime.h>
#include <hip/hip_bf16.h>

// ---------------------------------------------------------------------------
// LDS-LR forward for MI455X (gfx1250, wave32, WMMA 16x16x32 bf16)
//
//   Z   = U @ [B1 | M1']              (8192x1024)x(1024x384)  WMMA
//   Bu  = Z[:, :64] @ B2              (8192x64)x(64x1024)     WMMA
//   H   = chunked parallel diagonal scan over Bu (f32 carries)
//   R2  = H @ C1                      (8192x1024)x(1024x64)   WMMA
//   X   = [R2 | causal-shift(Z[:,64:])]
//   out = X @ [C2 ; M2']              (8192x384)x(384x1024)   WMMA, f32 out
// ---------------------------------------------------------------------------

typedef __attribute__((ext_vector_type(16))) __bf16        v16bf;
typedef __attribute__((ext_vector_type(8)))  float         v8f;
typedef __attribute__((ext_vector_type(16))) unsigned short v16u;
typedef __attribute__((ext_vector_type(8)))  unsigned short v8u;
typedef __attribute__((ext_vector_type(4)))  float         v4f;

#define T_SEQ 2048
#define BSZ   4
#define NST   1024
#define NIN   1024
#define NOUT  1024
#define RANK  64
#define KXW   5
#define NC1   384          // 64 + 64*5 fused columns
#define CHUNK 128
#define NCH   (T_SEQ / CHUNK)   // 16
#define MROWS (BSZ * T_SEQ)     // 8192

__device__ __forceinline__ unsigned short bf16_rn(float f) {
  unsigned u = __builtin_bit_cast(unsigned, f);
  u += 0x7FFFu + ((u >> 16) & 1u);          // round-to-nearest-even
  return (unsigned short)(u >> 16);
}
__device__ __forceinline__ float bf16_f32(unsigned short h) {
  unsigned u = ((unsigned)h) << 16;
  return __builtin_bit_cast(float, u);
}

// A fragment (16x32 bf16): lane<16 rows M=r, K = {half*8+0..7, 16+half*8+0..7}
__device__ __forceinline__ v16bf frag_a_bf16(const unsigned short* p, int half) {
  v8u lo = *(const v8u*)(p + half * 8);
  v8u hi = *(const v8u*)(p + 16 + half * 8);
  v16u c;
#pragma unroll
  for (int i = 0; i < 8; i++) { c[i] = lo[i]; c[i + 8] = hi[i]; }
  return __builtin_bit_cast(v16bf, c);
}
// B fragment from BT[N,K] row-major: lane gives N=r, K = {half*8+0..7, 16+half*8+0..7}
__device__ __forceinline__ v16bf frag_b_bf16(const unsigned short* p, int half) {
  v8u lo = *(const v8u*)(p + half * 8);
  v8u hi = *(const v8u*)(p + 16 + half * 8);
  v16u c;
#pragma unroll
  for (int i = 0; i < 8; i++) { c[i] = lo[i]; c[i + 8] = hi[i]; }
  return __builtin_bit_cast(v16bf, c);
}
// A fragment read directly from f32 memory, converted in-register
__device__ __forceinline__ v16bf frag_a_f32(const float* p, int half) {
  v4f x0 = *(const v4f*)(p + half * 8);
  v4f x1 = *(const v4f*)(p + half * 8 + 4);
  v4f y0 = *(const v4f*)(p + 16 + half * 8);
  v4f y1 = *(const v4f*)(p + 16 + half * 8 + 4);
  v16u c;
#pragma unroll
  for (int i = 0; i < 4; i++) {
    c[i]      = bf16_rn(x0[i]);
    c[i + 4]  = bf16_rn(x1[i]);
    c[i + 8]  = bf16_rn(y0[i]);
    c[i + 12] = bf16_rn(y1[i]);
  }
  return __builtin_bit_cast(v16bf, c);
}

__device__ __forceinline__ v8f wmma_bf16(v16bf a, v16bf b, v8f c) {
  return __builtin_amdgcn_wmma_f32_16x16x32_bf16(
      /*neg_a=*/false, a, /*neg_b=*/false, b,
      /*c_mod=*/(short)0, c, /*reuse_a=*/false, /*reuse_b=*/false);
}

// ---------------------------------------------------------------------------
// Generic bf16 GEMM: C[M,N] = A[M,K](lda) @ BT[N,K](ldb)^T
// block = 8 waves (4x2) of 32x32 tiles -> 128x64 per block.
// ---------------------------------------------------------------------------
__global__ __launch_bounds__(256) void gemm_bf16_kernel(
    const unsigned short* __restrict__ A, int lda,
    const unsigned short* __restrict__ BT, int ldb,
    void* __restrict__ C, int ldc, int K, int out_is_bf16) {
  const int lane = threadIdx.x & 31;
  const int wave = threadIdx.x >> 5;
  const int half = lane >> 4;
  const int r = lane & 15;
  const int m0 = blockIdx.y * 128 + (wave >> 1) * 32;
  const int n0 = blockIdx.x * 64 + (wave & 1) * 32;
  const unsigned short* a0p = A + (size_t)(m0 + r) * lda;
  const unsigned short* a1p = A + (size_t)(m0 + 16 + r) * lda;
  const unsigned short* b0p = BT + (size_t)(n0 + r) * ldb;
  const unsigned short* b1p = BT + (size_t)(n0 + 16 + r) * ldb;
  v8f acc00 = {}, acc01 = {}, acc10 = {}, acc11 = {};
  for (int kk = 0; kk < K; kk += 32) {
    v16bf a0 = frag_a_bf16(a0p + kk, half);
    v16bf a1 = frag_a_bf16(a1p + kk, half);
    v16bf b0 = frag_b_bf16(b0p + kk, half);
    v16bf b1 = frag_b_bf16(b1p + kk, half);
    acc00 = wmma_bf16(a0, b0, acc00);
    acc01 = wmma_bf16(a0, b1, acc01);
    acc10 = wmma_bf16(a1, b0, acc10);
    acc11 = wmma_bf16(a1, b1, acc11);
  }
#pragma unroll
  for (int v = 0; v < 8; v++) {
    const int row0 = m0 + v + half * 8;
    const int row1 = row0 + 16;
    const int col0 = n0 + r;
    const int col1 = col0 + 16;
    if (out_is_bf16) {
      unsigned short* Co = (unsigned short*)C;
      Co[(size_t)row0 * ldc + col0] = bf16_rn(acc00[v]);
      Co[(size_t)row0 * ldc + col1] = bf16_rn(acc01[v]);
      Co[(size_t)row1 * ldc + col0] = bf16_rn(acc10[v]);
      Co[(size_t)row1 * ldc + col1] = bf16_rn(acc11[v]);
    } else {
      float* Co = (float*)C;
      Co[(size_t)row0 * ldc + col0] = acc00[v];
      Co[(size_t)row0 * ldc + col1] = acc01[v];
      Co[(size_t)row1 * ldc + col0] = acc10[v];
      Co[(size_t)row1 * ldc + col1] = acc11[v];
    }
  }
}

// Input GEMM: A is f32 inputs (converted in-register), Z bf16 out.
// M=8192 K=1024 N=384.
__global__ __launch_bounds__(256) void gemm_in_kernel(
    const float* __restrict__ A, const unsigned short* __restrict__ BT,
    unsigned short* __restrict__ C) {
  const int lane = threadIdx.x & 31;
  const int wave = threadIdx.x >> 5;
  const int half = lane >> 4;
  const int r = lane & 15;
  const int m0 = blockIdx.y * 128 + (wave >> 1) * 32;
  const int n0 = blockIdx.x * 64 + (wave & 1) * 32;
  const float* a0p = A + (size_t)(m0 + r) * NIN;
  const float* a1p = A + (size_t)(m0 + 16 + r) * NIN;
  const unsigned short* b0p = BT + (size_t)(n0 + r) * NIN;
  const unsigned short* b1p = BT + (size_t)(n0 + 16 + r) * NIN;
  v8f acc00 = {}, acc01 = {}, acc10 = {}, acc11 = {};
  for (int kk = 0; kk < NIN; kk += 32) {
    v16bf a0 = frag_a_f32(a0p + kk, half);
    v16bf a1 = frag_a_f32(a1p + kk, half);
    v16bf b0 = frag_b_bf16(b0p + kk, half);
    v16bf b1 = frag_b_bf16(b1p + kk, half);
    acc00 = wmma_bf16(a0, b0, acc00);
    acc01 = wmma_bf16(a0, b1, acc01);
    acc10 = wmma_bf16(a1, b0, acc10);
    acc11 = wmma_bf16(a1, b1, acc11);
  }
#pragma unroll
  for (int v = 0; v < 8; v++) {
    const int row0 = m0 + v + half * 8;
    const int row1 = row0 + 16;
    const int col0 = n0 + r;
    const int col1 = col0 + 16;
    C[(size_t)row0 * NC1 + col0] = bf16_rn(acc00[v]);
    C[(size_t)row0 * NC1 + col1] = bf16_rn(acc01[v]);
    C[(size_t)row1 * NC1 + col0] = bf16_rn(acc10[v]);
    C[(size_t)row1 * NC1 + col1] = bf16_rn(acc11[v]);
  }
}

// ---------------------------------------------------------------------------
// Chunked parallel scan: h_t = A*h_{t-1} + Bu_t
// ---------------------------------------------------------------------------
// Phase 1: per-chunk local scan with zero init; emit chunk-end state S.
__global__ void scan_local_kernel(const unsigned short* __restrict__ Bu,
                                  const float* __restrict__ Adiag,
                                  unsigned short* __restrict__ Hloc,
                                  float* __restrict__ S) {
  const int g = blockIdx.x * blockDim.x + threadIdx.x;  // BSZ*NCH*NST
  const int n = g & (NST - 1);
  const int c = (g >> 10) & (NCH - 1);
  const int b = g >> 14;
  const float a = Adiag[n];
  float h = 0.f;
  const unsigned short* bu = Bu + ((size_t)(b * T_SEQ + c * CHUNK)) * NST + n;
  unsigned short* ho = Hloc + ((size_t)(b * T_SEQ + c * CHUNK)) * NST + n;
#pragma unroll 4
  for (int j = 0; j < CHUNK; j++) {
    __builtin_prefetch(bu + (size_t)(j + 8) * NST, 0, 1);  // global_prefetch_b8
    h = a * h + bf16_f32(bu[(size_t)j * NST]);
    ho[(size_t)j * NST] = bf16_rn(h);
  }
  S[(size_t)(b * NCH + c) * NST + n] = h;
}

// Phase 2: cross-chunk carry prefix (16 chunks; A^128 by 7 squarings).
__global__ void scan_carry_kernel(const float* __restrict__ S,
                                  const float* __restrict__ h0,
                                  const float* __restrict__ Adiag,
                                  float* __restrict__ Carry) {
  const int g = blockIdx.x * blockDim.x + threadIdx.x;  // BSZ*NST
  const int n = g & (NST - 1);
  const int b = g >> 10;
  const float a = Adiag[n];
  float aL = a;
#pragma unroll
  for (int i = 0; i < 7; i++) aL = aL * aL;  // a^128
  float carry = h0[n];
  Carry[(size_t)(b * NCH) * NST + n] = carry;
  for (int c = 1; c < NCH; c++) {
    carry = aL * carry + S[(size_t)(b * NCH + c - 1) * NST + n];
    Carry[(size_t)(b * NCH + c) * NST + n] = carry;
  }
}

// Phase 3: apply incoming carry: h_t += A^{j+1} * carry (in place).
__global__ void scan_fix_kernel(unsigned short* __restrict__ Hloc,
                                const float* __restrict__ Carry,
                                const float* __restrict__ Adiag) {
  const int g = blockIdx.x * blockDim.x + threadIdx.x;
  const int n = g & (NST - 1);
  const int c = (g >> 10) & (NCH - 1);
  const int b = g >> 14;
  const float a = Adiag[n];
  const float carry = Carry[(size_t)(b * NCH + c) * NST + n];
  unsigned short* ho = Hloc + ((size_t)(b * T_SEQ + c * CHUNK)) * NST + n;
  float p = a;
#pragma unroll 4
  for (int j = 0; j < CHUNK; j++) {
    float v = bf16_f32(ho[(size_t)j * NST]) + p * carry;
    ho[(size_t)j * NST] = bf16_rn(v);
    p *= a;
  }
}

// ---------------------------------------------------------------------------
// Weight packing: fused, transposed (so B operands stream K-contiguous b128s)
// WinT[n][i]:  n<64 -> B1[i][n];  n>=64 (q=n-64, r=q/5, k=q%5) -> M1[r][i][k]
// WoutT[o][c]: c<64 -> C2[c][o];  c>=64 -> M2[o][r][k]
// B2T[n][r] = B2[r][n];  C1T[r][n] = C1[n][r]
// ---------------------------------------------------------------------------
__global__ void pack_weights_kernel(const float* __restrict__ B1,
                                    const float* __restrict__ B2,
                                    const float* __restrict__ C1,
                                    const float* __restrict__ C2,
                                    const float* __restrict__ M1,
                                    const float* __restrict__ M2,
                                    unsigned short* __restrict__ WinT,
                                    unsigned short* __restrict__ B2T,
                                    unsigned short* __restrict__ C1T,
                                    unsigned short* __restrict__ WoutT) {
  const int g = blockIdx.x * blockDim.x + threadIdx.x;  // NC1*NIN threads
  if (g < NC1 * NIN) {
    const int nn = g / NIN, ii = g % NIN;
    float v;
    if (nn < RANK) v = B1[(size_t)ii * RANK + nn];
    else {
      const int q = nn - RANK;
      v = M1[((size_t)(q / KXW) * NIN + ii) * KXW + (q % KXW)];
    }
    WinT[g] = bf16_rn(v);
  }
  if (g < NOUT * NC1) {
    const int oo = g / NC1, cc = g % NC1;
    float v;
    if (cc < RANK) v = C2[(size_t)cc * NOUT + oo];
    else {
      const int q = cc - RANK;
      v = M2[((size_t)oo * RANK + (q / KXW)) * KXW + (q % KXW)];
    }
    WoutT[g] = bf16_rn(v);
  }
  if (g < NST * RANK) {
    const int nn = g / RANK, rr = g % RANK;
    B2T[g] = bf16_rn(B2[(size_t)rr * NST + nn]);
    const int r2 = g / NST, n2 = g % NST;
    C1T[g] = bf16_rn(C1[(size_t)n2 * RANK + r2]);
  }
}

// X = [R2 | causal-shift of Z AR columns]; column c>=64 shifted by k=(c-64)%5
__global__ void pack_x_kernel(const unsigned short* __restrict__ R2,
                              const unsigned short* __restrict__ Z,
                              unsigned short* __restrict__ X) {
  const int g = blockIdx.x * blockDim.x + threadIdx.x;  // MROWS*NC1
  if (g >= MROWS * NC1) return;
  const int bt = g / NC1;
  const int c = g % NC1;
  unsigned short v;
  if (c < RANK) {
    v = R2[(size_t)bt * RANK + c];
  } else {
    const int k = (c - RANK) % KXW;
    const int t = bt & (T_SEQ - 1);
    v = (t >= k) ? Z[(size_t)(bt - k) * NC1 + c] : (unsigned short)0;
  }
  X[g] = v;
}

// ---------------------------------------------------------------------------
extern "C" void kernel_launch(void* const* d_in, const int* in_sizes, int n_in,
                              void* d_out, int out_size, void* d_ws, size_t ws_size,
                              hipStream_t stream) {
  (void)in_sizes; (void)n_in; (void)out_size; (void)ws_size;
  const float* inputs = (const float*)d_in[0];
  const float* h0     = (const float*)d_in[1];
  const float* Adiag  = (const float*)d_in[2];
  const float* B1     = (const float*)d_in[3];
  const float* B2     = (const float*)d_in[4];
  const float* C1     = (const float*)d_in[5];
  const float* C2     = (const float*)d_in[6];
  const float* M1     = (const float*)d_in[7];
  const float* M2     = (const float*)d_in[8];
  float* out = (float*)d_out;

  char* ws = (char*)d_ws;  // ~47.3 MB layout
  unsigned short* Zbf   = (unsigned short*)(ws + 0);          // 8192x384 bf16
  unsigned short* Bu    = (unsigned short*)(ws + 6291456);    // 8192x1024 bf16
  unsigned short* Hloc  = (unsigned short*)(ws + 23068672);   // 8192x1024 bf16
  float*          S     = (float*)(ws + 39845888);            // 4x16x1024 f32
  float*          Carry = (float*)(ws + 40108032);            // 4x16x1024 f32
  unsigned short* R2    = (unsigned short*)(ws + 40370176);   // 8192x64 bf16
  unsigned short* X     = (unsigned short*)(ws + 41418752);   // 8192x384 bf16
  unsigned short* WinT  = (unsigned short*)(ws + 47710208);   // 384x1024 bf16
  unsigned short* B2T   = (unsigned short*)(ws + 48496640);   // 1024x64 bf16
  unsigned short* C1T   = (unsigned short*)(ws + 48627712);   // 64x1024 bf16
  unsigned short* WoutT = (unsigned short*)(ws + 48758784);   // 1024x384 bf16

  // 1) pack/transpose weights to bf16
  pack_weights_kernel<<<(NC1 * NIN) / 256, 256, 0, stream>>>(
      B1, B2, C1, C2, M1, M2, WinT, B2T, C1T, WoutT);
  // 2) Z = U @ [B1|M1']  (f32 A converted in-register)
  gemm_in_kernel<<<dim3(NC1 / 64, MROWS / 128), 256, 0, stream>>>(inputs, WinT, Zbf);
  // 3) Bu = Z[:, :64] @ B2
  gemm_bf16_kernel<<<dim3(NST / 64, MROWS / 128), 256, 0, stream>>>(
      Zbf, NC1, B2T, RANK, Bu, NST, RANK, 1);
  // 4-6) chunked parallel scan
  scan_local_kernel<<<(BSZ * NCH * NST) / 256, 256, 0, stream>>>(Bu, Adiag, Hloc, S);
  scan_carry_kernel<<<(BSZ * NST) / 256, 256, 0, stream>>>(S, h0, Adiag, Carry);
  scan_fix_kernel<<<(BSZ * NCH * NST) / 256, 256, 0, stream>>>(Hloc, Carry, Adiag);
  // 7) R2 = H @ C1
  gemm_bf16_kernel<<<dim3(RANK / 64, MROWS / 128), 256, 0, stream>>>(
      Hloc, NST, C1T, NST, R2, RANK, NST, 1);
  // 8) X = [R2 | shifted AR projections]
  pack_x_kernel<<<(MROWS * NC1) / 256, 256, 0, stream>>>(R2, Zbf, X);
  // 9) out = X @ [C2; M2']  (f32 output)
  gemm_bf16_kernel<<<dim3(NOUT / 64, MROWS / 128), 256, 0, stream>>>(
      X, NC1, WoutT, NC1, out, NOUT, NC1, 0);
}
